// GroupAttention_64115271794696
// MI455X (gfx1250) — matrix operand order repaired and checked by reading, not compiled, not run on hardware
//
#include <hip/hip_runtime.h>
#include <hip/hip_bf16.h>

// Problem constants
#define BB 4
#define SS 2048
#define DD 1024
#define BS (BB * SS)          // 8192 rows
#define DSCALE 512.0f         // d_model // 2

typedef __attribute__((ext_vector_type(16))) __bf16 v16bf;
typedef __attribute__((ext_vector_type(8)))  float  v8f;
typedef __attribute__((ext_vector_type(4)))  __bf16 v4bf;

struct alignas(16) u32x4 { unsigned int x, y, z, w; };
union Frag16 { v16bf v; u32x4 q[2]; };

// ---------------------------------------------------------------------------
// Kernel 1: LayerNorm (fp32 in) -> bf16 normalized activations.
// One 256-thread block per row; 4 elements/thread; wave32 shfl + LDS reduce.
// ---------------------------------------------------------------------------
__global__ void __launch_bounds__(256) ln_bf16_kernel(
    const float* __restrict__ ctx, const float* __restrict__ gamma,
    const float* __restrict__ beta, __bf16* __restrict__ xo) {
  const int row = blockIdx.x;
  const int t = threadIdx.x;
  const float4 v = ((const float4*)(ctx + (size_t)row * DD))[t];
  float s  = v.x + v.y + v.z + v.w;
  float ss = v.x * v.x + v.y * v.y + v.z * v.z + v.w * v.w;
  #pragma unroll
  for (int off = 16; off > 0; off >>= 1) {
    s  += __shfl_down(s,  off, 32);
    ss += __shfl_down(ss, off, 32);
  }
  __shared__ float rs[8], rss[8];
  const int wid = t >> 5, lane = t & 31;
  if (lane == 0) { rs[wid] = s; rss[wid] = ss; }
  __syncthreads();
  float S1 = 0.f, S2 = 0.f;
  #pragma unroll
  for (int i = 0; i < 8; ++i) { S1 += rs[i]; S2 += rss[i]; }
  const float mu   = S1 * (1.0f / DD);
  const float var  = S2 * (1.0f / DD) - mu * mu;
  const float rstd = rsqrtf(var + 1e-5f);
  const float4 gv = ((const float4*)gamma)[t];
  const float4 bv = ((const float4*)beta)[t];
  v4bf o;
  o[0] = (__bf16)((v.x - mu) * rstd * gv.x + bv.x);
  o[1] = (__bf16)((v.y - mu) * rstd * gv.y + bv.y);
  o[2] = (__bf16)((v.z - mu) * rstd * gv.z + bv.z);
  o[3] = (__bf16)((v.w - mu) * rstd * gv.w + bv.w);
  ((v4bf*)(xo + (size_t)row * DD))[t] = o;
}

// ---------------------------------------------------------------------------
// Kernel 2: convert wq, wk (fp32, torch Linear layout [out,in]) to
// bf16 in the same [out(N), in(K)] layout (no transpose); B fragments are two 16B loads per lane.
// ---------------------------------------------------------------------------
__global__ void __launch_bounds__(256) convw_kernel(
    const float* __restrict__ wq, const float* __restrict__ wk,
    __bf16* __restrict__ wqT, __bf16* __restrict__ wkT) {
  const int idx = blockIdx.x * 256 + threadIdx.x;   // idx over output, same [out,in] layout as input
  const int k = idx >> 10, n = idx & (DD - 1);
  wqT[idx] = (__bf16)wq[k * DD + n];
  wkT[idx] = (__bf16)wk[k * DD + n];
}

// ---------------------------------------------------------------------------
// Kernel 3: fused Q/K projection GEMM via v_wmma_f32_16x16x32_bf16.
//   Q = X @ Wq^T + bq ; K = X @ Wk^T + bk     (M=8192, N=1024, K=1024)
// One wave per 16x16 output tile, both Q and K accumulators share A fragments
// (2x A reuse). Weight tiles (4 MB bf16 total) stay L2-resident across all
// 512 M-tiles; the 192 MB L2 makes this compute-bound on the WMMA pipe.
// Fragment layouts follow CDNA5 ISA §7.12.2 exactly:
//   A 16x32 bf16: lane m=L&15, half=L>>4; elems 0-7 -> k=kb+8*half+j,
//                 elems 8-15 -> k=kb+16+8*half+(j-8)  (two 16B chunks)
//   B 32x16 bf16: lane col n=n0+(L&15), half=L>>4; elems use A's k order (two 16B chunks)
//   C 16x16 f32 : lane col = L&15, vgpr v -> row m0 + v + 8*(L>>4)
// ---------------------------------------------------------------------------
__global__ void __launch_bounds__(256) gemm_qk_kernel(
    const __bf16* __restrict__ X, const __bf16* __restrict__ WqT,
    const __bf16* __restrict__ WkT, const float* __restrict__ bq,
    const float* __restrict__ bk, float* __restrict__ Q,
    float* __restrict__ K) {
  const int lane = threadIdx.x & 31;
  const int wave = threadIdx.x >> 5;
  const int tile = blockIdx.x * 8 + wave;   // 512*64 = 32768 tiles
  const int tn = tile & 63;                 // N/16
  const int tm = tile >> 6;                 // M/16
  const int m0 = tm * 16, n0 = tn * 16;

  const int am = lane & 15, ah = lane >> 4;
  const __bf16* aRow  = X + (size_t)(m0 + am) * DD;
  const __bf16* bqRow = WqT + (size_t)(n0 + am) * DD;  // n = n0 + (lane & 15)
  const __bf16* bkRow = WkT + (size_t)(n0 + am) * DD;

  v8f accq = {0.f, 0.f, 0.f, 0.f, 0.f, 0.f, 0.f, 0.f};
  v8f acck = {0.f, 0.f, 0.f, 0.f, 0.f, 0.f, 0.f, 0.f};

  for (int kb = 0; kb < DD; kb += 32) {
    __builtin_prefetch(aRow + kb + 128, 0, 0);  // global_prefetch_b8 on A stream
    Frag16 a, b1, b2;
    a.q[0] = *(const u32x4*)(aRow + kb + 8 * ah);
    a.q[1] = *(const u32x4*)(aRow + kb + 16 + 8 * ah);
    const __bf16* bp1 = bqRow + kb;
    b1.q[0] = *(const u32x4*)(bp1 + 8 * ah);
    b1.q[1] = *(const u32x4*)(bp1 + 16 + 8 * ah);
    const __bf16* bp2 = bkRow + kb;
    b2.q[0] = *(const u32x4*)(bp2 + 8 * ah);
    b2.q[1] = *(const u32x4*)(bp2 + 16 + 8 * ah);
    accq = __builtin_amdgcn_wmma_f32_16x16x32_bf16(
        false, a.v, false, b1.v, (short)0, accq, false, false);
    acck = __builtin_amdgcn_wmma_f32_16x16x32_bf16(
        false, a.v, false, b2.v, (short)0, acck, false, false);
  }

  const int cl = lane & 15, ch = lane >> 4;
  const float biq = bq[n0 + cl];
  const float bik = bk[n0 + cl];
  #pragma unroll
  for (int v = 0; v < 8; ++v) {
    const int row = m0 + v + 8 * ch;
    Q[(size_t)row * DD + n0 + cl] = accq[v] + biq;
    K[(size_t)row * DD + n0 + cl] = acck[v] + bik;
  }
}

// ---------------------------------------------------------------------------
// Kernel 4: per-row neighbor scores + masked 2-way softmax.
// Mask keeps only j=i±1 where eos_mask[b,j]!=0; everything else is -1e9, and
// exp(-1e9 - m) underflows to exactly 0 in fp32 -> closed-form softmax.
// Rows with no valid neighbor become uniform 1/S (flagged).
// One wave per row; 1024-long dot products q[i]·k[i±1].
// ---------------------------------------------------------------------------
__global__ void __launch_bounds__(256) row_scores_kernel(
    const float* __restrict__ Q, const float* __restrict__ K,
    const int* __restrict__ eos, float* __restrict__ pup,
    float* __restrict__ pdn, float* __restrict__ unf) {
  const int lane = threadIdx.x & 31;
  const int r = blockIdx.x * 8 + (threadIdx.x >> 5);
  const int i = r & (SS - 1);
  const bool hu = (i + 1) < SS, hd = i > 0;
  const float4* qr = (const float4*)(Q + (size_t)r * DD);
  const float4* ku = (const float4*)(K + (size_t)(r + 1) * DD);
  const float4* kd = (const float4*)(K + (size_t)(r - 1) * DD);
  float du = 0.f, dd = 0.f;
  for (int c = lane; c < DD / 4; c += 32) {
    const float4 qv = qr[c];
    if (hu) { const float4 u = ku[c]; du += qv.x*u.x + qv.y*u.y + qv.z*u.z + qv.w*u.w; }
    if (hd) { const float4 d = kd[c]; dd += qv.x*d.x + qv.y*d.y + qv.z*d.z + qv.w*d.w; }
  }
  #pragma unroll
  for (int off = 16; off > 0; off >>= 1) {
    du += __shfl_down(du, off, 32);
    dd += __shfl_down(dd, off, 32);
  }
  if (lane == 0) {
    const float su = du * (1.0f / DSCALE);
    const float sd = dd * (1.0f / DSCALE);
    const bool vu = hu && (eos[r + 1] != 0);
    const bool vd = hd && (eos[r - 1] != 0);
    float pu = 0.f, pd = 0.f, un = 0.f;
    if (vu && vd) {
      const float m = fmaxf(su, sd);
      const float eu = expf(su - m), ed = expf(sd - m);
      const float Z = eu + ed;
      pu = eu / Z; pd = ed / Z;
    } else if (vu) pu = 1.f;
    else if (vd) pd = 1.f;
    else un = 1.f;
    pup[r] = pu; pdn[r] = pd; unf[r] = un;
  }
}

// ---------------------------------------------------------------------------
// Kernel 5: superdiagonal of transformed neibor -> l[i]=log(nb+1e-9) -> prefix
// sums P per batch (g[i,j] = exp(P[max]-P[min]) + 1e-9). 4 serial scans of
// 2048 elements — microseconds, not worth a parallel scan.
// ---------------------------------------------------------------------------
__global__ void prefix_kernel(const float* __restrict__ pup,
                              const float* __restrict__ pdn,
                              const float* __restrict__ unf,
                              const float* __restrict__ prior_p,
                              float* __restrict__ P) {
  const int b = threadIdx.x;
  if (b >= BB) return;
  const float prior = prior_p[0];
  const float invS = 1.0f / SS;
  const int base = b * SS;
  float run = 0.f;
  P[base] = 0.f;
  for (int i = 0; i < SS - 1; ++i) {
    const float n_ij = unf[base + i]     > 0.5f ? invS : pup[base + i];
    const float n_ji = unf[base + i + 1] > 0.5f ? invS : pdn[base + i + 1];
    const float nb = prior + (1.f - prior) * sqrtf(n_ij * n_ji + 1e-9f);
    run += logf(nb + 1e-9f);
    P[base + i + 1] = run;
  }
}

// ---------------------------------------------------------------------------
// Kernel 6: materialize g and neibor (134 MB of stores -> HBM-bound floor).
// All per-row metadata (128 KB) is L2/L0 resident; writes fully coalesced.
// ---------------------------------------------------------------------------
__global__ void __launch_bounds__(256) writer_kernel(
    const float* __restrict__ pup, const float* __restrict__ pdn,
    const float* __restrict__ unf, const float* __restrict__ P,
    const float* __restrict__ prior_p, float* __restrict__ g_out,
    float* __restrict__ nb_out) {
  const int r = blockIdx.x;                 // global row in [0, B*S)
  const int b = r >> 11;
  const int i = r & (SS - 1);
  const float prior = prior_p[0];
  const float one_m = 1.f - prior;
  const float invS = 1.0f / SS;
  const bool ui = unf[r] > 0.5f;
  const float pu_i = pup[r], pd_i = pdn[r];
  const float Pi = P[r];
  const size_t rowOff = (size_t)r * SS;
  const int base = b * SS;
  for (int j = threadIdx.x; j < SS; j += 256) {
    const int rj = base + j;
    const bool uj = unf[rj] > 0.5f;
    float n_ij, n_ji;
    if (j == i + 1)      { n_ij = ui ? invS : pu_i; n_ji = uj ? invS : pdn[rj]; }
    else if (j == i - 1) { n_ij = ui ? invS : pd_i; n_ji = uj ? invS : pup[rj]; }
    else                 { n_ij = ui ? invS : 0.f;  n_ji = uj ? invS : 0.f; }
    const float nb = prior + one_m * sqrtf(n_ij * n_ji + 1e-9f);
    float gv;
    if (j == i) {
      gv = nb;                               // diagonal: neibor(i,i)
    } else {
      const float Pj = P[rj];
      const float dlt = (j > i) ? (Pj - Pi) : (Pi - Pj);
      gv = expf(dlt) + 1e-9f;
    }
    g_out[rowOff + j]  = gv;
    nb_out[rowOff + j] = nb;
  }
}

// ---------------------------------------------------------------------------
extern "C" void kernel_launch(void* const* d_in, const int* in_sizes, int n_in,
                              void* d_out, int out_size, void* d_ws, size_t ws_size,
                              hipStream_t stream) {
  const float* ctx   = (const float*)d_in[0];
  const int*   eos   = (const int*)d_in[1];
  const float* prior = (const float*)d_in[2];
  const float* wq    = (const float*)d_in[3];
  const float* bq    = (const float*)d_in[4];
  const float* wk    = (const float*)d_in[5];
  const float* bk    = (const float*)d_in[6];
  const float* gamma = (const float*)d_in[7];
  const float* beta  = (const float*)d_in[8];

  char* ws = (char*)d_ws;
  __bf16* xbf = (__bf16*)ws;                                  // 16.78 MB
  __bf16* wqT = (__bf16*)(ws + (size_t)16777216);             //  2.10 MB
  __bf16* wkT = (__bf16*)(ws + (size_t)16777216 + 2097152);   //  2.10 MB
  float*  Qm  = (float*)(ws + (size_t)16777216 + 4194304);    // 33.55 MB
  float*  Km  = Qm + (size_t)BS * DD;                         // 33.55 MB
  float*  pup = Km + (size_t)BS * DD;
  float*  pdn = pup + BS;
  float*  unf = pdn + BS;
  float*  P   = unf + BS;

  float* g_out  = (float*)d_out;
  float* nb_out = g_out + (size_t)BB * SS * SS;

  ln_bf16_kernel<<<BS, 256, 0, stream>>>(ctx, gamma, beta, xbf);
  convw_kernel<<<(DD * DD) / 256, 256, 0, stream>>>(wq, wk, wqT, wkT);
  gemm_qk_kernel<<<(512 * 64) / 8, 256, 0, stream>>>(xbf, wqT, wkT, bq, bk, Qm, Km);
  row_scores_kernel<<<BS / 8, 256, 0, stream>>>(Qm, Km, eos, pup, pdn, unf);
  prefix_kernel<<<1, 32, 0, stream>>>(pup, pdn, unf, prior, P);
  writer_kernel<<<BS, 256, 0, stream>>>(pup, pdn, unf, P, prior, g_out, nb_out);
}
